// NeuralFingerprint_33062658244851
// MI455X (gfx1250) — matrix operand order repaired and checked by reading, not compiled, not run on hardware
//
#include <hip/hip_runtime.h>
#include <hip/hip_bf16.h>

typedef __bf16 bf16_t;
typedef __attribute__((ext_vector_type(8)))  __bf16 v8bf;
typedef __attribute__((ext_vector_type(16))) __bf16 v16bf;
typedef __attribute__((ext_vector_type(8)))  float  v8f;

#define F_DIM     128
#define D_BUCKETS 11
#define FP_DIM    1024

// ---------------------------------------------------------------------------
// gfx1250 async global->LDS copy (ASYNCcnt path), with safe fallback.
#if __has_builtin(__builtin_amdgcn_global_load_async_to_lds_b128)
typedef int v4i_t __attribute__((vector_size(16)));
typedef __attribute__((address_space(1))) v4i_t* gv4i_p;   // global (__device__) int4*
typedef __attribute__((address_space(3))) v4i_t* lv4i_p;   // LDS (__shared__) int4*
__device__ inline void async_copy16(const void* g, void* l) {
  // (global int4* src, lds int4* dst, imm offset, imm cpol)
  __builtin_amdgcn_global_load_async_to_lds_b128(
      (gv4i_p)(unsigned long long)g,
      (lv4i_p)(unsigned)(unsigned long long)l, 0, 0);
}
__device__ inline void async_wait_all() {
#if __has_builtin(__builtin_amdgcn_s_wait_asynccnt)
  __builtin_amdgcn_s_wait_asynccnt(0);
#else
  asm volatile("s_wait_asynccnt 0x0" ::: "memory");
#endif
}
#else
__device__ inline void async_copy16(const void* g, void* l) {
  *(uint4*)l = *(const uint4*)g;               // sync load + ds_store fallback
}
__device__ inline void async_wait_all() {}
#endif

// ---------------------------------------------------------------------------
// WMMA fragment loaders (CDNA5 16-bit layouts, ISA 7.12.2)
// A matrix 16x32 (MxK): lane L holds row M=L&15.
//   lanes 0-15 : elems 0..7 = K 0..7,  elems 8..15 = K 16..23
//   lanes 16-31: elems 0..7 = K 8..15, elems 8..15 = K 24..31
__device__ inline v16bf frag_a(const bf16_t* rowbase, int kbase) {
  int lane = threadIdx.x & 31;
  int half = lane >> 4;
  const bf16_t* p = rowbase + kbase + half * 8;
  v8bf lo = *(const v8bf*)p;
  v8bf hi = *(const v8bf*)(p + 16);
  v16bf f;
#pragma unroll
  for (int i = 0; i < 8; ++i) { f[i] = lo[i]; f[i + 8] = hi[i]; }
  return f;
}

// B matrix 32x16 (KxN): lane L holds column N=L&15 (weight rows are out-major,
// in-contiguous => a weight row IS a B column slice).
//   same K order as A: lanes 0-15 hold K 0..7,16..23; lanes 16-31 hold K 8..15,24..31.
__device__ inline v16bf frag_b(const bf16_t* rowbase, int kbase) {
  int lane = threadIdx.x & 31;
  int half = lane >> 4;
  const bf16_t* p = rowbase + kbase + half * 8;
  v8bf lo = *(const v8bf*)p;
  v8bf hi = *(const v8bf*)(p + 16);
  v16bf f;
#pragma unroll
  for (int i = 0; i < 8; ++i) { f[i] = lo[i]; f[i + 8] = hi[i]; }
  return f;
}

__device__ inline v8f wmma_bf16(v16bf a, v16bf b, v8f c) {
  return __builtin_amdgcn_wmma_f32_16x16x32_bf16(false, a, false, b, (short)0, c,
                                                 false, false);
}

// ---------------------------------------------------------------------------
__global__ void f32_to_bf16_k(const float* __restrict__ in, bf16_t* __restrict__ out,
                              long n) {
  long i = (long)blockIdx.x * 256 + threadIdx.x;
  if (i < n) out[i] = (bf16_t)in[i];
}

// Fused: h[dst] += x[src] (f32 accumulate from bf16) and incoming-degree count
__global__ void agg_kernel(const bf16_t* __restrict__ x_bf, const int* __restrict__ src,
                           const int* __restrict__ dst, float* __restrict__ h,
                           int* __restrict__ deg, int E) {
  long tid = (long)blockIdx.x * 256 + threadIdx.x;
  long e = tid >> 5;
  if (e >= E) return;
  int q = (int)(tid & 31);            // 32 threads/edge, 4 features each
  int s = src[e], d = dst[e];
  if (q == 0) atomicAdd(&deg[d], 1);
  const bf16_t* xp = x_bf + (long)s * F_DIM + q * 4;
  float* hp = h + (long)d * F_DIM + q * 4;
#pragma unroll
  for (int i = 0; i < 4; ++i) atomicAdd(hp + i, (float)xp[i]);
}

__global__ void bucket_count(const int* __restrict__ deg, int* __restrict__ cnt, int N) {
  int i = blockIdx.x * 256 + threadIdx.x;
  if (i < N) atomicAdd(&cnt[min(deg[i], D_BUCKETS - 1)], 1);
}

// Exclusive scan over 11 buckets, tiles padded to 16 rows; fills tile->bucket map
__global__ void bucket_scan(const int* __restrict__ cnt, int* __restrict__ offb,
                            int* __restrict__ tile_bucket) {
  if (threadIdx.x == 0 && blockIdx.x == 0) {
    int off = 0;
    for (int d = 0; d < D_BUCKETS; ++d) {
      offb[d] = off;
      int nt = (cnt[d] + 15) >> 4;
      int t0 = off >> 4;
      for (int t = 0; t < nt; ++t) tile_bucket[t0 + t] = d;
      off += nt << 4;
    }
  }
}

__global__ void bucket_scatter(const int* __restrict__ deg, const int* __restrict__ offb,
                               int* __restrict__ fill, int* __restrict__ perm, int N) {
  int i = blockIdx.x * 256 + threadIdx.x;
  if (i >= N) return;
  int d = min(deg[i], D_BUCKETS - 1);
  int pos = offb[d] + atomicAdd(&fill[d], 1);
  perm[pos] = i;
}

// ---------------------------------------------------------------------------
// MFConv GEMM: per 16-node tile (single degree bucket d):
//   out = sigmoid( h_tile @ Wl[d]^T + x_tile @ Wr[d]^T + bl[d] )
// A tiles (h rows, x rows) staged once per block into LDS via async copy and
// shared by all 8 waves; each wave owns one 16x16 output feature tile.
__global__ __launch_bounds__(256) void mfconv_wmma(
    const bf16_t* __restrict__ h_bf, const bf16_t* __restrict__ x_bf,
    const bf16_t* __restrict__ Wl_bf, const bf16_t* __restrict__ Wr_bf,
    const float* __restrict__ bl, const int* __restrict__ perm,
    const int* __restrict__ tile_bucket, bf16_t* __restrict__ x_out) {
  __shared__ alignas(16) bf16_t hA[16 * F_DIM];   // 4 KB
  __shared__ alignas(16) bf16_t xA[16 * F_DIM];   // 4 KB
  int t = blockIdx.x;
  int d = tile_bucket[t];
  if (d < 0) return;                       // unused (padded) tile: uniform exit
  int tid = threadIdx.x;

  // Stage the 16x128 A tiles: thread -> (row = tid>>4, 16B chunk = tid&15)
  {
    int row = tid >> 4, chunk = tid & 15;
    int g = perm[t * 16 + row];
    int loff = row * F_DIM + chunk * 8;
    if (g >= 0) {
      long goff = (long)g * F_DIM + chunk * 8;
      async_copy16(h_bf + goff, &hA[loff]);
      async_copy16(x_bf + goff, &xA[loff]);
    } else {                               // padded row -> zeros
      uint4 z = {0u, 0u, 0u, 0u};
      *(uint4*)&hA[loff] = z;
      *(uint4*)&xA[loff] = z;
    }
  }
  async_wait_all();
  __syncthreads();

  int wave = tid >> 5;                     // output feature tile 0..7
  int lane = tid & 31;
  int half = lane >> 4;
  int r = lane & 15;

  const bf16_t* hrow = &hA[r * F_DIM];
  const bf16_t* xrow = &xA[r * F_DIM];
  const bf16_t* wl = Wl_bf + ((long)d * F_DIM + wave * 16 + r) * F_DIM;
  const bf16_t* wr = Wr_bf + ((long)d * F_DIM + wave * 16 + r) * F_DIM;

  v8f acc;
#pragma unroll
  for (int i = 0; i < 8; ++i) acc[i] = 0.0f;
#pragma unroll
  for (int kk = 0; kk < F_DIM; kk += 32) {
    acc = wmma_bf16(frag_a(hrow, kk), frag_b(wl, kk), acc);
    acc = wmma_bf16(frag_a(xrow, kk), frag_b(wr, kk), acc);
  }

  int n = wave * 16 + r;                   // output feature index
  float bias = bl[d * F_DIM + n];
#pragma unroll
  for (int v = 0; v < 8; ++v) {            // C layout: M = v + 8*half, N = lane&15
    int m = v + half * 8;
    int gm = perm[t * 16 + m];
    if (gm >= 0) {
      float val = acc[v] + bias;
      float sg = 1.0f / (1.0f + __expf(-val));
      x_out[(long)gm * F_DIM + n] = (bf16_t)sg;
    }
  }
}

// ---------------------------------------------------------------------------
// Linear 128->1024 + row softmax + column-sum readout (atomic into fp[1024]).
__global__ __launch_bounds__(256) void linear_softmax_wmma(
    const bf16_t* __restrict__ x_bf, const bf16_t* __restrict__ Wlin_bf,
    const float* __restrict__ blin, float* __restrict__ fp_out, int N) {
  __shared__ alignas(16) bf16_t zsh[16 * FP_DIM];   // 32 KB logits/probs tile
  __shared__ alignas(16) bf16_t xA[16 * F_DIM];     // 4 KB A tile
  int t = blockIdx.x;
  int tid = threadIdx.x;

  // Stage the 16x128 node-feature tile once per block
  {
    int row = tid >> 4, chunk = tid & 15;
    int node = t * 16 + row;
    int loff = row * F_DIM + chunk * 8;
    if (node < N) {
      async_copy16(x_bf + (long)node * F_DIM + chunk * 8, &xA[loff]);
    } else {
      uint4 z = {0u, 0u, 0u, 0u};
      *(uint4*)&xA[loff] = z;
    }
  }
  async_wait_all();
  __syncthreads();

  int wave = tid >> 5;
  int lane = tid & 31;
  int half = lane >> 4;
  int r = lane & 15;

  const bf16_t* xrow = &xA[r * F_DIM];
  v16bf a0 = frag_a(xrow, 0);
  v16bf a1 = frag_a(xrow, 32);
  v16bf a2 = frag_a(xrow, 64);
  v16bf a3 = frag_a(xrow, 96);

  for (int nt = wave; nt < FP_DIM / 16; nt += 8) {
    const bf16_t* wrow = Wlin_bf + (long)(nt * 16 + r) * F_DIM;
    v8f acc;
#pragma unroll
    for (int i = 0; i < 8; ++i) acc[i] = 0.0f;
    acc = wmma_bf16(a0, frag_b(wrow, 0), acc);
    acc = wmma_bf16(a1, frag_b(wrow, 32), acc);
    acc = wmma_bf16(a2, frag_b(wrow, 64), acc);
    acc = wmma_bf16(a3, frag_b(wrow, 96), acc);
    int n = nt * 16 + r;
    float bias = blin[n];
#pragma unroll
    for (int v = 0; v < 8; ++v) {
      int m = v + half * 8;
      zsh[m * FP_DIM + n] = (bf16_t)(acc[v] + bias);
    }
  }
  __syncthreads();

  // Row softmax: wave w owns rows 2w (lanes 0-15) and 2w+1 (lanes 16-31);
  // half-wave xor-shuffle reductions over the 16 lanes of each half.
  int row = 2 * wave + half;
  float mx = -1e30f;
  for (int j = r; j < FP_DIM; j += 16) mx = fmaxf(mx, (float)zsh[row * FP_DIM + j]);
#pragma unroll
  for (int o = 1; o < 16; o <<= 1) mx = fmaxf(mx, __shfl_xor(mx, o, 32));
  float s = 0.0f;
  for (int j = r; j < FP_DIM; j += 16) s += __expf((float)zsh[row * FP_DIM + j] - mx);
#pragma unroll
  for (int o = 1; o < 16; o <<= 1) s += __shfl_xor(s, o, 32);
  float inv = 1.0f / s;
  for (int j = r; j < FP_DIM; j += 16) {
    float e = __expf((float)zsh[row * FP_DIM + j] - mx) * inv;
    zsh[row * FP_DIM + j] = (bf16_t)e;
  }
  __syncthreads();

  // Column partial sums over the block's 16 rows -> one global atomic per column
  for (int c = tid; c < FP_DIM; c += 256) {
    float s2 = 0.0f;
#pragma unroll
    for (int m = 0; m < 16; ++m)
      if (t * 16 + m < N) s2 += (float)zsh[m * FP_DIM + c];
    atomicAdd(&fp_out[c], s2);
  }
}

// ---------------------------------------------------------------------------
extern "C" void kernel_launch(void* const* d_in, const int* in_sizes, int n_in,
                              void* d_out, int out_size, void* d_ws, size_t ws_size,
                              hipStream_t stream) {
  const float* x_in  = (const float*)d_in[0];
  const int*   ei    = (const int*)d_in[1];
  const float* Wl    = (const float*)d_in[2];
  const float* bl    = (const float*)d_in[3];
  const float* Wr    = (const float*)d_in[4];
  const float* Wlin  = (const float*)d_in[5];
  const float* blin  = (const float*)d_in[6];

  const int N  = in_sizes[0] / F_DIM;
  const int E  = in_sizes[1] / 2;
  const int L  = in_sizes[2] / (D_BUCKETS * F_DIM * F_DIM);
  const int FP = in_sizes[5] / (L * F_DIM);      // == FP_DIM

  const int* src = ei;
  const int* dst = ei + E;

  // Workspace carve (256B aligned)
  char* wsp = (char*)d_ws;
  size_t off = 0;
  auto carve = [&](size_t bytes) -> void* {
    void* p = wsp + off;
    off += (bytes + 255) & ~(size_t)255;
    return p;
  };
  const long nXF = (long)N * F_DIM;
  bf16_t* xbf0 = (bf16_t*)carve(nXF * 2);
  bf16_t* xbf1 = (bf16_t*)carve(nXF * 2);
  bf16_t* hbf  = (bf16_t*)carve(nXF * 2);
  float*  hacc = (float*)carve(nXF * 4);
  int*    deg  = (int*)carve((size_t)N * 4);
  int*    cnt  = (int*)carve(16 * 4);
  int*    offb = (int*)carve(16 * 4);
  int*    fill = (int*)carve(16 * 4);
  const int np = N + D_BUCKETS * 16;             // padded perm length
  int*    perm = (int*)carve((size_t)np * 4);
  const int nt_max = (N + 15) / 16 + D_BUCKETS;  // max padded tiles
  int*    tb   = (int*)carve((size_t)nt_max * 4);
  bf16_t* WlB  = (bf16_t*)carve((size_t)D_BUCKETS * F_DIM * F_DIM * 2);
  bf16_t* WrB  = (bf16_t*)carve((size_t)D_BUCKETS * F_DIM * F_DIM * 2);
  bf16_t* WlnB = (bf16_t*)carve((size_t)FP * F_DIM * 2);

  (void)hipMemsetAsync(d_out, 0, (size_t)FP * sizeof(float), stream);

  // Layer-0 features to bf16
  f32_to_bf16_k<<<(unsigned)((nXF + 255) / 256), 256, 0, stream>>>(x_in, xbf0, nXF);

  bf16_t* xcur  = xbf0;
  bf16_t* xnext = xbf1;

  for (int l = 0; l < L; ++l) {
    const float* Wl_l   = Wl   + (long)l * D_BUCKETS * F_DIM * F_DIM;
    const float* bl_l   = bl   + (long)l * D_BUCKETS * F_DIM;
    const float* Wr_l   = Wr   + (long)l * D_BUCKETS * F_DIM * F_DIM;
    const float* Wlin_l = Wlin + (long)l * FP * F_DIM;
    const float* blin_l = blin + (long)l * FP;

    const long nW = (long)D_BUCKETS * F_DIM * F_DIM;
    f32_to_bf16_k<<<(unsigned)((nW + 255) / 256), 256, 0, stream>>>(Wl_l, WlB, nW);
    f32_to_bf16_k<<<(unsigned)((nW + 255) / 256), 256, 0, stream>>>(Wr_l, WrB, nW);
    const long nWL = (long)FP * F_DIM;
    f32_to_bf16_k<<<(unsigned)((nWL + 255) / 256), 256, 0, stream>>>(Wlin_l, WlnB, nWL);

    (void)hipMemsetAsync(hacc, 0, (size_t)nXF * 4, stream);
    (void)hipMemsetAsync(deg, 0, (size_t)N * 4, stream);
    (void)hipMemsetAsync(cnt, 0, 16 * 4, stream);
    (void)hipMemsetAsync(fill, 0, 16 * 4, stream);
    (void)hipMemsetAsync(perm, 0xFF, (size_t)np * 4, stream);       // -1
    (void)hipMemsetAsync(tb, 0xFF, (size_t)nt_max * 4, stream);     // -1

    agg_kernel<<<(unsigned)(((long)E * 32 + 255) / 256), 256, 0, stream>>>(
        xcur, src, dst, hacc, deg, E);
    bucket_count<<<(unsigned)((N + 255) / 256), 256, 0, stream>>>(deg, cnt, N);
    bucket_scan<<<1, 1, 0, stream>>>(cnt, offb, tb);
    bucket_scatter<<<(unsigned)((N + 255) / 256), 256, 0, stream>>>(deg, offb, fill,
                                                                    perm, N);
    f32_to_bf16_k<<<(unsigned)((nXF + 255) / 256), 256, 0, stream>>>(hacc, hbf, nXF);

    mfconv_wmma<<<(unsigned)nt_max, 256, 0, stream>>>(hbf, xcur, WlB, WrB, bl_l, perm,
                                                      tb, xnext);
    linear_softmax_wmma<<<(unsigned)((N + 15) / 16), 256, 0, stream>>>(
        xnext, WlnB, blin_l, (float*)d_out, N);

    bf16_t* tmp = xcur; xcur = xnext; xnext = tmp;
  }
}
